// EncoderLayer_89412629168400
// MI455X (gfx1250) — matrix operand order repaired and checked by reading, not compiled, not run on hardware
//
#include <hip/hip_runtime.h>
#include <hip/hip_bf16.h>

#define S_LEN 1024
#define D_DIM 512
#define H_NUM 8
#define HS_DIM 64
#define B_NUM 16
#define DFF_DIM 2048
#define TOKENS (B_NUM * S_LEN)   // 16384

typedef __attribute__((ext_vector_type(16))) __bf16 bf16x16;
typedef __attribute__((ext_vector_type(8)))  float  v8f;

union FragU { bf16x16 v; uint4 q[2]; };

__device__ inline v8f wmma_bf16(bf16x16 a, bf16x16 b, v8f c) {
  return __builtin_amdgcn_wmma_f32_16x16x32_bf16(
      false, a, false, b, (short)0, c, false, false);
}

// A-matrix 16x32 fragment (row-major source, lane = M, K chunks half*8 & 16+half*8)
__device__ inline bf16x16 load_frag_a(const __bf16* base, int ld) {
  int lane = threadIdx.x & 31;
  int r = lane & 15, hf = lane >> 4;
  const __bf16* p = base + (size_t)r * ld + hf * 8;
  FragU u;
  u.q[0] = *reinterpret_cast<const uint4*>(p);
  u.q[1] = *reinterpret_cast<const uint4*>(p + 16);
  return u.v;
}

// B-matrix 32x16 fragment from transposed source Bt[N][K] (lane = N, K chunks half*8 & 16+half*8)
__device__ inline bf16x16 load_frag_b(const __bf16* base, int ld) {
  int lane = threadIdx.x & 31;
  int n = lane & 15, hf = lane >> 4;
  const __bf16* p = base + (size_t)n * ld + hf * 8;
  FragU u;
  u.q[0] = *reinterpret_cast<const uint4*>(p);
  u.q[1] = *reinterpret_cast<const uint4*>(p + 16);
  return u.v;
}

// ---------------- pack / convert kernels ----------------
__global__ void pack_qkv_kernel(const float* qk, const float* kk, const float* vk,
                                __bf16* out) {
  int idx = blockIdx.x * 256 + threadIdx.x;       // 1536*512
  int n = idx >> 9, i = idx & 511;
  int sel = n >> 9, c = n & 511;
  int h = c >> 6, o = c & 63;
  size_t src = ((size_t)h * D_DIM + i) * HS_DIM + o;
  float v;
  if (sel == 0)      v = qk[src] * 0.125f;        // fold 1/sqrt(64) into Wq
  else if (sel == 1) v = kk[src];
  else               v = vk[src];
  out[idx] = (__bf16)v;
}

__global__ void pack_proj_kernel(const float* proj, __bf16* out) {
  int idx = blockIdx.x * 256 + threadIdx.x;       // 512*512 ; out[o][h*64+i]
  int o = idx >> 9, c = idx & 511;
  out[idx] = (__bf16)proj[(size_t)c * D_DIM + o];
}

__global__ void pack_w1_kernel(const float* w1, __bf16* out) {
  int idx = blockIdx.x * 256 + threadIdx.x;       // 2048*512 ; out[n][i]
  int n = idx >> 9, i = idx & 511;
  out[idx] = (__bf16)w1[(size_t)i * DFF_DIM + n];
}

__global__ void pack_w2_kernel(const float* w2, __bf16* out) {
  int idx = blockIdx.x * 256 + threadIdx.x;       // 512*2048 ; out[n][j]
  int n = idx >> 11, j = idx & 2047;
  out[idx] = (__bf16)w2[(size_t)j * D_DIM + n];
}

__global__ void convert_x_kernel(const float* x, __bf16* out) {
  size_t idx = (size_t)blockIdx.x * 256 + threadIdx.x;  // 16384*512
  out[idx] = (__bf16)x[idx];
}

// ---------------- generic bf16 WMMA GEMM ----------------
// A: [M][K] row-major bf16.  Bt: [N][K] (transposed B) bf16.
// Block = 256 threads = 8 waves (2x4), wave tile 32x64, block tile 64x256.
// MODE 0: QKV scatter (N=1536); MODE 1: +bias+resid -> f32 (ld 512);
// MODE 2: +bias, ReLU -> bf16 (ld 2048)
template <int MODE>
__global__ void gemm_bf16_kernel(const __bf16* __restrict__ A,
                                 const __bf16* __restrict__ Bt, int K,
                                 const float* __restrict__ bias,
                                 const float* __restrict__ resid,
                                 float* __restrict__ outf,
                                 __bf16* __restrict__ outb,
                                 __bf16* __restrict__ Qp,
                                 __bf16* __restrict__ Kp,
                                 __bf16* __restrict__ Vtp) {
  int w = threadIdx.x >> 5;
  int wm = w >> 2, wn = w & 3;
  int m0 = blockIdx.x * 64 + wm * 32;
  int n0 = blockIdx.y * 256 + wn * 64;
  int lane = threadIdx.x & 31, nl = lane & 15, hf = lane >> 4;

  v8f acc[2][4] = {};
  for (int k = 0; k < K; k += 32) {
    bf16x16 a0 = load_frag_a(A + (size_t)m0 * K + k, K);
    bf16x16 a1 = load_frag_a(A + (size_t)(m0 + 16) * K + k, K);
#pragma unroll
    for (int j = 0; j < 4; ++j) {
      bf16x16 bf = load_frag_b(Bt + (size_t)(n0 + j * 16) * K + k, K);
      acc[0][j] = wmma_bf16(a0, bf, acc[0][j]);
      acc[1][j] = wmma_bf16(a1, bf, acc[1][j]);
    }
  }

#pragma unroll
  for (int i = 0; i < 2; ++i) {
#pragma unroll
    for (int j = 0; j < 4; ++j) {
#pragma unroll
      for (int r = 0; r < 8; ++r) {
        int m = m0 + i * 16 + r + hf * 8;
        int n = n0 + j * 16 + nl;
        float v = acc[i][j][r];
        if (MODE == 0) {
          int sel = n >> 9, c = n & 511;
          int h = c >> 6, hs = c & 63;
          int b = m >> 10, s = m & 1023;
          __bf16 bv = (__bf16)v;
          size_t rowmaj = (((size_t)b * H_NUM + h) * S_LEN + s) * HS_DIM + hs;
          if (sel == 0)      Qp[rowmaj] = bv;
          else if (sel == 1) Kp[rowmaj] = bv;
          else Vtp[(((size_t)b * H_NUM + h) * HS_DIM + hs) * S_LEN + s] = bv;
        } else if (MODE == 1) {
          size_t o = (size_t)m * D_DIM + n;
          outf[o] = v + bias[n] + resid[o];
        } else {  // MODE 2
          outb[(size_t)m * DFF_DIM + n] = (__bf16)fmaxf(v + bias[n], 0.0f);
        }
      }
    }
  }
}

// ---------------- flash attention ----------------
// Q,K: [B,H,S,HS] bf16.  Vt: [B,H,HS,S] bf16.  O: [B,S,H*HS] bf16.
// 256 threads = 8 waves; each wave owns 16 query rows; grid = B*H*(S/128).
__global__ void attn_kernel(const __bf16* __restrict__ Q,
                            const __bf16* __restrict__ Kmat,
                            const __bf16* __restrict__ Vt,
                            const unsigned char* __restrict__ mask,
                            const float* __restrict__ td,
                            const float* __restrict__ coef,
                            __bf16* __restrict__ O) {
  __shared__ __align__(16) __bf16 Pst[8][16][32];
  int blk = blockIdx.x;
  int bh = blk >> 3, qt = blk & 7;
  int b = bh >> 3, h = bh & 7;
  int w = threadIdx.x >> 5;
  int lane = threadIdx.x & 31, nl = lane & 15, hf = lane >> 4;
  int q0 = qt * 128 + w * 16;

  const __bf16* Qb = Q + (((size_t)bh) * S_LEN + q0) * HS_DIM;
  const __bf16* Kb = Kmat + ((size_t)bh) * S_LEN * HS_DIM;
  const __bf16* Vb = Vt + ((size_t)bh) * HS_DIM * S_LEN;
  float ch = coef[h];

  bf16x16 aq0 = load_frag_a(Qb, HS_DIM);
  bf16x16 aq1 = load_frag_a(Qb + 32, HS_DIM);

  v8f acc[4] = {};
  float mrun[8], lrun[8];
#pragma unroll
  for (int r = 0; r < 8; ++r) { mrun[r] = -1e30f; lrun[r] = 0.0f; }

  for (int kt = 0; kt < S_LEN; kt += 32) {
    v8f ct[2];
#pragma unroll
    for (int jt = 0; jt < 2; ++jt) {
      int kb = kt + jt * 16;
      bf16x16 bk0 = load_frag_b(Kb + (size_t)kb * HS_DIM, HS_DIM);
      bf16x16 bk1 = load_frag_b(Kb + (size_t)kb * HS_DIM + 32, HS_DIM);
      v8f c = {};
      c = wmma_bf16(aq0, bk0, c);
      c = wmma_bf16(aq1, bk1, c);
      int key = kb + nl;
      const unsigned char* mrow = mask + ((size_t)b * S_LEN + q0) * S_LEN + key;
      const float* trow = td + ((size_t)b * S_LEN + q0) * S_LEN + key;
#pragma unroll
      for (int r = 0; r < 8; ++r) {
        int row = r + hf * 8;
        float lg = c[r];
        if (!mrow[(size_t)row * S_LEN]) lg -= 1e10f;
        lg -= trow[(size_t)row * S_LEN] * ch;
        c[r] = lg;
      }
      ct[jt] = c;
    }
    // online softmax over this 32-key block
#pragma unroll
    for (int r = 0; r < 8; ++r) {
      int row = r + hf * 8;
      float lg0 = ct[0][r], lg1 = ct[1][r];
      float mx = fmaxf(lg0, lg1);
#pragma unroll
      for (int o = 1; o < 16; o <<= 1) mx = fmaxf(mx, __shfl_xor(mx, o, 32));
      float mn = fmaxf(mrun[r], mx);
      float sc = __expf(mrun[r] - mn);
      float p0 = __expf(lg0 - mn), p1 = __expf(lg1 - mn);
      float rs = p0 + p1;
#pragma unroll
      for (int o = 1; o < 16; o <<= 1) rs += __shfl_xor(rs, o, 32);
      lrun[r] = lrun[r] * sc + rs;
      mrun[r] = mn;
#pragma unroll
      for (int jn = 0; jn < 4; ++jn) acc[jn][r] *= sc;
      Pst[w][row][nl] = (__bf16)p0;
      Pst[w][row][16 + nl] = (__bf16)p1;
    }
    asm volatile("s_wait_dscnt 0x0" ::: "memory");  // wave-private LDS RAW
    bf16x16 pa = load_frag_a(&Pst[w][0][0], 32);
#pragma unroll
    for (int jn = 0; jn < 4; ++jn) {
      bf16x16 bv = load_frag_b(Vb + (size_t)(jn * 16) * S_LEN + kt, S_LEN);
      acc[jn] = wmma_bf16(pa, bv, acc[jn]);
    }
  }

#pragma unroll
  for (int jn = 0; jn < 4; ++jn)
#pragma unroll
    for (int r = 0; r < 8; ++r) {
      int row = r + hf * 8;
      float o = acc[jn][r] / fmaxf(lrun[r], 1e-30f);
      O[((size_t)b * S_LEN + q0 + row) * D_DIM + h * HS_DIM + jn * 16 + nl] =
          (__bf16)o;
    }
}

// ---------------- layernorm (per row of 512) ----------------
__global__ void ln_kernel(const float* __restrict__ x,
                          const float* __restrict__ g,
                          const float* __restrict__ bta,
                          float* __restrict__ outf,
                          __bf16* __restrict__ outb) {
  __shared__ float rs[8], rs2[8];
  int row = blockIdx.x, t = threadIdx.x;
  const float* xr = x + (size_t)row * D_DIM;
  float e0 = xr[t], e1 = xr[t + 256];
  float s = e0 + e1, s2 = e0 * e0 + e1 * e1;
#pragma unroll
  for (int o = 1; o < 32; o <<= 1) {
    s += __shfl_xor(s, o, 32);
    s2 += __shfl_xor(s2, o, 32);
  }
  int w = t >> 5, lane = t & 31;
  if (lane == 0) { rs[w] = s; rs2[w] = s2; }
  __syncthreads();
  if (w == 0) {
    float a = (lane < 8) ? rs[lane] : 0.0f;
    float a2 = (lane < 8) ? rs2[lane] : 0.0f;
#pragma unroll
    for (int o = 1; o < 8; o <<= 1) {
      a += __shfl_xor(a, o, 32);
      a2 += __shfl_xor(a2, o, 32);
    }
    if (lane == 0) { rs[0] = a; rs2[0] = a2; }
  }
  __syncthreads();
  float mu = rs[0] * (1.0f / D_DIM);
  float var = rs2[0] * (1.0f / D_DIM) - mu * mu;
  float rstd = rsqrtf(var + 1e-6f);
  float y0 = (e0 - mu) * rstd * g[t] + bta[t];
  float y1 = (e1 - mu) * rstd * g[t + 256] + bta[t + 256];
  outf[(size_t)row * D_DIM + t] = y0;
  outf[(size_t)row * D_DIM + t + 256] = y1;
  if (outb) {
    outb[(size_t)row * D_DIM + t] = (__bf16)y0;
    outb[(size_t)row * D_DIM + t + 256] = (__bf16)y1;
  }
}

extern "C" void kernel_launch(void* const* d_in, const int* in_sizes, int n_in,
                              void* d_out, int out_size, void* d_ws, size_t ws_size,
                              hipStream_t stream) {
  (void)in_sizes; (void)n_in; (void)out_size; (void)ws_size;
  const float* value   = (const float*)d_in[0];
  const unsigned char* mask = (const unsigned char*)d_in[1];
  const float* timedif = (const float*)d_in[2];
  const float* coefw   = (const float*)d_in[3];
  const float* q_k     = (const float*)d_in[4];
  const float* k_k     = (const float*)d_in[5];
  const float* v_k     = (const float*)d_in[6];
  const float* proj_k  = (const float*)d_in[7];
  const float* proj_b  = (const float*)d_in[8];
  const float* ln1_g   = (const float*)d_in[9];
  const float* ln1_b   = (const float*)d_in[10];
  const float* w1      = (const float*)d_in[11];
  const float* b1      = (const float*)d_in[12];
  const float* w2      = (const float*)d_in[13];
  const float* b2      = (const float*)d_in[14];
  const float* ln2_g   = (const float*)d_in[15];
  const float* ln2_b   = (const float*)d_in[16];
  float* outp = (float*)d_out;

  char* ws = (char*)d_ws;
  size_t off = 0;
  auto alloc = [&](size_t bytes) -> void* {
    void* p = ws + off;
    off += (bytes + 255) & ~(size_t)255;
    return p;
  };
  __bf16* Wqkv  = (__bf16*)alloc((size_t)1536 * 512 * 2);
  __bf16* Wproj = (__bf16*)alloc((size_t)512 * 512 * 2);
  __bf16* W1t   = (__bf16*)alloc((size_t)2048 * 512 * 2);
  __bf16* W2t   = (__bf16*)alloc((size_t)512 * 2048 * 2);
  __bf16* Xb    = (__bf16*)alloc((size_t)TOKENS * D_DIM * 2);   // reused as MH
  __bf16* Qw    = (__bf16*)alloc((size_t)TOKENS * D_DIM * 2);   // + next = preLN1
  __bf16* Kw    = (__bf16*)alloc((size_t)TOKENS * D_DIM * 2);
  __bf16* Vtw   = (__bf16*)alloc((size_t)TOKENS * D_DIM * 2);   // reused as out1b
  float*  out1  = (float*)alloc((size_t)TOKENS * D_DIM * 4);
  __bf16* Hws   = (__bf16*)alloc((size_t)TOKENS * DFF_DIM * 2);
  (void)Kw;
  float*  preLN1 = (float*)Qw;   // 32MB spanning Qw+Kw, free after attention
  __bf16* MH     = Xb;           // Xb free after QKV GEMM
  __bf16* out1b  = Vtw;          // Vt free after attention

  // 1. pack weights + convert activations to bf16
  pack_qkv_kernel<<<3072, 256, 0, stream>>>(q_k, k_k, v_k, Wqkv);
  pack_proj_kernel<<<1024, 256, 0, stream>>>(proj_k, Wproj);
  pack_w1_kernel<<<4096, 256, 0, stream>>>(w1, W1t);
  pack_w2_kernel<<<4096, 256, 0, stream>>>(w2, W2t);
  convert_x_kernel<<<TOKENS * D_DIM / 256, 256, 0, stream>>>(value, Xb);

  // 2. fused QKV projection (N = 1536)
  gemm_bf16_kernel<0><<<dim3(TOKENS / 64, 6), 256, 0, stream>>>(
      Xb, Wqkv, D_DIM, nullptr, nullptr, nullptr, nullptr, Qw, Kw, Vtw);

  // 3. flash attention -> MH [tokens, 512] bf16
  attn_kernel<<<B_NUM * H_NUM * (S_LEN / 128), 256, 0, stream>>>(
      Qw, Kw, Vtw, mask, timedif, coefw, MH);

  // 4. output projection + bias + residual -> preLN1 (f32)
  gemm_bf16_kernel<1><<<dim3(TOKENS / 64, 2), 256, 0, stream>>>(
      MH, Wproj, D_DIM, proj_b, value, preLN1, nullptr, nullptr, nullptr, nullptr);

  // 5. layernorm 1 -> out1 (f32) + out1b (bf16)
  ln_kernel<<<TOKENS, 256, 0, stream>>>(preLN1, ln1_g, ln1_b, out1, out1b);

  // 6. FFN1 + ReLU -> Hws bf16 [tokens, 2048]
  gemm_bf16_kernel<2><<<dim3(TOKENS / 64, 8), 256, 0, stream>>>(
      out1b, W1t, D_DIM, b1, nullptr, nullptr, Hws, nullptr, nullptr, nullptr);

  // 7. FFN2 + bias + residual -> d_out (pre-LN2, f32)
  gemm_bf16_kernel<1><<<dim3(TOKENS / 64, 2), 256, 0, stream>>>(
      Hws, W2t, DFF_DIM, b2, out1, outp, nullptr, nullptr, nullptr, nullptr);

  // 8. layernorm 2 in-place on d_out
  ln_kernel<<<TOKENS, 256, 0, stream>>>(outp, ln2_g, ln2_b, outp, nullptr);
}
